// HGLEnet_50852412785461
// MI455X (gfx1250) — matrix operand order repaired and checked by reading, not compiled, not run on hardware
//
#include <hip/hip_runtime.h>
#include <hip/hip_bf16.h>

// ---------------------------------------------------------------------------
// HGLEnet forward for MI455X (gfx1250).
// Heavy matmuls (temporal convs as implicit GEMM, 1x1 conv, feat@feat^T,
// adj@feat, feat-GEMM vs [entry_w|pool_w]) use V_WMMA_F32_16X16X32_F16.
// Small graph-pooling / head stages are VALU kernels. f32 accumulation
// throughout; f16 only for WMMA operands.
// ---------------------------------------------------------------------------

typedef __attribute__((ext_vector_type(16))) _Float16 v16h;
typedef __attribute__((ext_vector_type(8)))  float    v8f;

constexpr int NB   = 32;     // batch
constexpr int NC   = 62;     // channels (graph nodes)
constexpr int NTT  = 1024;   // time samples
constexpr int NTF  = 32;     // temporal filters
constexpr int LTOT = 708;    // concat pooled length (247+240+221)
constexpr int LHALF= 354;    // after AvgPool(1,2)
constexpr int FDIM = 11328;  // 32*354
constexpr int OGn  = 64;
constexpr int PGn  = 32;

__device__ __forceinline__ v8f wmma16(v16h a, v16h b, v8f c) {
  return __builtin_amdgcn_wmma_f32_16x16x32_f16(
      /*neg_a=*/false, a, /*neg_b=*/false, b, /*c_mod=*/(short)0, c,
      /*reuse_a=*/false, /*reuse_b=*/false);
}
// A-fragment K index for half h (0..15):
// lanes 0-15 hold K {0..15}; lanes 16-31 hold K {16..31}
__device__ __forceinline__ int akm(int h, int hi) {
  return h + (hi ? 16 : 0);
}

// ===========================================================================
// S1: 3-branch temporal conv (implicit GEMM, f16 WMMA) + square + AvgPool16/4
//     + log  ->  out0 [NB, NTF, NC, LTOT]
// ===========================================================================
__global__ void k_conv(const float* __restrict__ x,
                       const float* __restrict__ w1, const float* __restrict__ b1,
                       const float* __restrict__ w2, const float* __restrict__ b2,
                       const float* __restrict__ w3, const float* __restrict__ b3,
                       float* __restrict__ out0) {
  const int KS_[3] = {25, 51, 128};
  const int KP_[3] = {32, 64, 128};
  const int LL_[3] = {1000, 974, 897};
  const int LP_[3] = {247, 240, 221};
  const int TO_[3] = {0, 247, 487};

  extern __shared__ char smem[];
  _Float16* xs  = (_Float16*)smem;        // 1152 padded input row (f16)
  _Float16* wsm = xs + 1152;              // 32*128 padded weights (f16)
  float*    bsm = (float*)(wsm + 32*128); // 32 biases
  float*    ysq = bsm + 32;               // 32*1008 conv^2 staging

  int br = blockIdx.y;
  int bc = blockIdx.x;
  int b = bc / NC, c = bc % NC;
  const float* wp = br == 0 ? w1 : (br == 1 ? w2 : w3);
  const float* bp = br == 0 ? b1 : (br == 1 ? b2 : b3);
  int k = KS_[br], Kp = KP_[br], L = LL_[br], Lp = LP_[br];
  int tid = threadIdx.x;

  for (int t = tid; t < 1152; t += blockDim.x) {
    float v = (t < NTT) ? x[(b * NC + c) * NTT + t] : 0.f;
    xs[t] = (_Float16)v;
  }
  for (int i = tid; i < NTF * Kp; i += blockDim.x) {
    int o = i / Kp, j = i % Kp;
    wsm[i] = (_Float16)((j < k) ? wp[o * k + j] : 0.f);
  }
  if (tid < NTF) bsm[tid] = bp[tid];
  __syncthreads();

  int wv = tid / 32, lane = tid % 32, hi = lane >> 4, ln = lane & 15;
  int ntiles = (L + 15) / 16;
  for (int nt = wv; nt < ntiles; nt += 8) {          // wave-uniform loop: EXEC all-1s
    int l0 = nt * 16;
    v8f acc0 = {}; v8f acc1 = {};
    for (int kc = 0; kc < Kp; kc += 32) {
      v16h a0, a1, bm;
#pragma unroll
      for (int h = 0; h < 16; ++h) {
        int ka = akm(h, hi);
        a0[h] = wsm[ln * Kp + kc + ka];              // A rows o=0..15
        a1[h] = wsm[(ln + 16) * Kp + kc + ka];       // A rows o=16..31
        bm[h] = xs[l0 + ln + kc + hi * 16 + h];      // B[k,n] = x[l0+n+j]
      }
      acc0 = wmma16(a0, bm, acc0);
      acc1 = wmma16(a1, bm, acc1);
    }
    int l = l0 + ln;
    if (l < L) {
#pragma unroll
      for (int r = 0; r < 8; ++r) {
        int o0 = r + hi * 8;
        float y0 = acc0[r] + bsm[o0];
        ysq[o0 * 1008 + l] = y0 * y0;
        float y1 = acc1[r] + bsm[o0 + 16];
        ysq[(o0 + 16) * 1008 + l] = y1 * y1;
      }
    }
  }
  __syncthreads();
  for (int i = tid; i < NTF * Lp; i += blockDim.x) {
    int o = i / Lp, p = i % Lp;
    float s = 0.f;
#pragma unroll 4
    for (int q = 0; q < 16; ++q) s += ysq[o * 1008 + 4 * p + q];
    out0[((b * NTF + o) * NC + c) * LTOT + TO_[br] + p] = logf(s * (1.f / 16.f));
  }
}

// ===========================================================================
// Generic per-channel batchnorm stats: src viewed as [outer, mid, inner],
// block m reduces over (outer, inner) -> mean[m], istd[m].
// ===========================================================================
__global__ void k_stats(const float* __restrict__ src, int outer, int mid, int inner,
                        float* __restrict__ mean, float* __restrict__ istd) {
  int m = blockIdx.x;
  float s = 0.f, q = 0.f;
  int n = outer * inner;
  for (int i = threadIdx.x; i < n; i += blockDim.x) {
    int o = i / inner, f = i % inner;
    float v = src[(o * mid + m) * inner + f];
    s += v; q += v * v;
  }
  __shared__ float rs[256], rq[256];
  rs[threadIdx.x] = s; rq[threadIdx.x] = q;
  __syncthreads();
  for (int st = 128; st > 0; st >>= 1) {
    if (threadIdx.x < st) { rs[threadIdx.x] += rs[threadIdx.x + st];
                            rq[threadIdx.x] += rq[threadIdx.x + st]; }
    __syncthreads();
  }
  if (threadIdx.x == 0) {
    float inv = 1.f / (float)n;
    float mu = rs[0] * inv;
    float var = rq[0] * inv - mu * mu;
    mean[m] = mu; istd[m] = rsqrtf(var + 1e-5f);
  }
}

// ===========================================================================
// S3: BN_t folded into 1x1 conv weights (WMMA), leaky-relu, AvgPool(1,2)
//     out0 [NB,32,NC,708] -> out1 [NB,32,NC,354]
// ===========================================================================
__global__ void k_oxo(const float* __restrict__ out0,
                      const float* __restrict__ oxo_w, const float* __restrict__ oxo_b,
                      const float* __restrict__ m0, const float* __restrict__ is0,
                      float* __restrict__ out1) {
  extern __shared__ char smem[];
  _Float16* y0h = (_Float16*)smem;          // 32*720
  _Float16* wph = y0h + 32 * 720;           // 32*32 folded weights
  float*    cb  = (float*)(wph + 32 * 32);  // 32 folded bias
  float*    zb  = cb + 32;                  // 32*712 post-leaky staging

  int bc = blockIdx.x; int b = bc / NC, c = bc % NC;
  int tid = threadIdx.x;
  for (int i = tid; i < 32 * 32; i += blockDim.x) {
    int o = i % 32;
    wph[i] = (_Float16)(oxo_w[i] * is0[o]);  // fold BN scale into W
  }
  if (tid < 32) {
    float s = oxo_b[tid];
    for (int o = 0; o < 32; ++o) s -= oxo_w[tid * 32 + o] * m0[o] * is0[o];
    cb[tid] = s;                              // fold BN mean into bias
  }
  for (int i = tid; i < 32 * 720; i += blockDim.x) {
    int o = i / 720, p = i % 720;
    y0h[i] = (_Float16)(p < LTOT ? out0[((b * NTF + o) * NC + c) * LTOT + p] : 0.f);
  }
  __syncthreads();

  int wv = tid / 32, lane = tid % 32, hi = lane >> 4, ln = lane & 15;
  for (int nt = wv; nt < 45; nt += 8) {
    int l0 = nt * 16;
    v16h a0, a1, bm;
#pragma unroll
    for (int h = 0; h < 16; ++h) {
      int ka = akm(h, hi);
      a0[h] = wph[ln * 32 + ka];
      a1[h] = wph[(ln + 16) * 32 + ka];
      bm[h] = y0h[(hi * 16 + h) * 720 + l0 + ln];
    }
    v8f acc0 = {}; v8f acc1 = {};
    acc0 = wmma16(a0, bm, acc0);
    acc1 = wmma16(a1, bm, acc1);
    int l = l0 + ln;
    if (l < LTOT) {
#pragma unroll
      for (int r = 0; r < 8; ++r) {
        int o0 = r + hi * 8;
        float z0 = acc0[r] + cb[o0];
        zb[o0 * 712 + l] = z0 > 0.f ? z0 : 0.01f * z0;
        float z1 = acc1[r] + cb[o0 + 16];
        zb[(o0 + 16) * 712 + l] = z1 > 0.f ? z1 : 0.01f * z1;
      }
    }
  }
  __syncthreads();
  for (int i = tid; i < NTF * LHALF; i += blockDim.x) {
    int oo = i / LHALF, p = i % LHALF;
    out1[((b * NTF + oo) * NC + c) * LHALF + p] =
        0.5f * (zb[oo * 712 + 2 * p] + zb[oo * 712 + 2 * p + 1]);
  }
}

// ===========================================================================
// S5: features = relu(BN(out1)*lf_w - lf_b)  ->  feat_pre [NB, NC, FDIM]
// ===========================================================================
__global__ void k_feat(const float* __restrict__ out1,
                       const float* __restrict__ m1, const float* __restrict__ is1,
                       const float* __restrict__ lf_w, const float* __restrict__ lf_b,
                       float* __restrict__ feat_pre) {
  int bc = blockIdx.x; int b = bc / NC, c = bc % NC;
  for (int f = threadIdx.x; f < FDIM; f += blockDim.x) {
    int oo = f / LHALF, p = f % LHALF;
    float v = (out1[((b * NTF + oo) * NC + c) * LHALF + p] - m1[oo]) * is1[oo];
    v = v * lf_w[c * FDIM + f] - lf_b[c];
    feat_pre[bc * FDIM + f] = fmaxf(v, 0.f);
  }
}

// normalize feat (BN over (0,2)) -> f16 [NB, 64(pad), FDIM], rows 62..63 = 0
__global__ void k_featnorm(const float* __restrict__ fp,
                           const float* __restrict__ mc, const float* __restrict__ isc,
                           _Float16* __restrict__ fh) {
  int total = NB * 64 * FDIM;
  for (int i = blockIdx.x * blockDim.x + threadIdx.x; i < total;
       i += gridDim.x * blockDim.x) {
    int f = i % FDIM; int r = (i / FDIM) % 64; int b = i / (FDIM * 64);
    float v = 0.f;
    if (r < NC) v = (fp[(b * NC + r) * FDIM + f] - mc[r]) * isc[r];
    fh[i] = (_Float16)v;
  }
}

// ===========================================================================
// S6: s = feat@feat^T (WMMA, K=11328); adj = relu(s*(G+G^T)) + I; sym-norm.
//     One block per batch; 16 waves = 4x4 tiles of 16.
// ===========================================================================
__global__ void k_adj(const _Float16* __restrict__ fh, const float* __restrict__ gadj,
                      float* __restrict__ adjf, _Float16* __restrict__ adjh) {
  __shared__ float ap[64 * 64];
  __shared__ float dsh[64];
  int b = blockIdx.x, tid = threadIdx.x;
  int wv = tid / 32, lane = tid % 32, hi = lane >> 4, ln = lane & 15;
  int ti = wv / 4, tj = wv % 4;
  const _Float16* fb = fh + (size_t)b * 64 * FDIM;
  v8f acc = {};
  for (int kc = 0; kc < FDIM; kc += 32) {
    v16h a, bm;
#pragma unroll
    for (int h = 0; h < 16; ++h) {
      a[h]  = fb[(ti * 16 + ln) * FDIM + kc + akm(h, hi)];
      bm[h] = fb[(tj * 16 + ln) * FDIM + kc + hi * 16 + h];  // B[k,n]=feat[n,k]
    }
    acc = wmma16(a, bm, acc);
  }
#pragma unroll
  for (int r = 0; r < 8; ++r) {
    int m = ti * 16 + r + hi * 8;
    int n = tj * 16 + ln;
    float v = 0.f;
    if (m < NC && n < NC) {
      float g = gadj[m * NC + n] + gadj[n * NC + m];
      v = fmaxf(acc[r] * g, 0.f);
      if (m == n) v += 1.f;
    }
    ap[m * 64 + n] = v;
  }
  __syncthreads();
  if (tid < 64) {
    float s = 0.f;
    for (int j = 0; j < 64; ++j) s += ap[tid * 64 + j];
    if (s == 0.f) s = 1.f;
    dsh[tid] = rsqrtf(s);
  }
  __syncthreads();
  for (int i = tid; i < 64 * 64; i += blockDim.x) {
    int m = i / 64, n = i % 64;
    float v = ap[i] * dsh[m] * dsh[n];
    adjf[b * 4096 + i] = v;
    adjh[b * 4096 + i] = (_Float16)v;
  }
}

// ===========================================================================
// S7: h_pre = adj @ feat (WMMA, M=64 K=64 N=11328) -> h16 [1984, FDIM] f16
// ===========================================================================
__global__ void k_hpre(const _Float16* __restrict__ adjh, const _Float16* __restrict__ fh,
                       _Float16* __restrict__ h16) {
  int n0 = blockIdx.x * 16;
  int b = blockIdx.y;
  int tid = threadIdx.x;
  int ti = tid / 32, lane = tid % 32, hi = lane >> 4, ln = lane & 15;
  v8f acc = {};
  for (int kc = 0; kc < 64; kc += 32) {
    v16h a, bm;
#pragma unroll
    for (int h = 0; h < 16; ++h) {
      a[h]  = adjh[b * 4096 + (ti * 16 + ln) * 64 + kc + akm(h, hi)];
      bm[h] = fh[((size_t)b * 64 + kc + hi * 16 + h) * FDIM + n0 + ln];
    }
    acc = wmma16(a, bm, acc);
  }
#pragma unroll
  for (int r = 0; r < 8; ++r) {
    int m = ti * 16 + r + hi * 8;
    if (m < NC) h16[(size_t)(b * NC + m) * FDIM + n0 + ln] = (_Float16)acc[r];
  }
}

// pack [entry_w | pool_w] -> f16 [FDIM, 96]
__global__ void k_wcat(const float* __restrict__ entry_w, const float* __restrict__ pool_w,
                       _Float16* __restrict__ wcat) {
  int total = FDIM * 96;
  for (int i = blockIdx.x * blockDim.x + threadIdx.x; i < total;
       i += gridDim.x * blockDim.x) {
    int k = i / 96, j = i % 96;
    wcat[i] = (_Float16)(j < OGn ? entry_w[k * OGn + j] : pool_w[k * PGn + (j - OGn)]);
  }
}

// ===========================================================================
// S8: Y = h16[1984,11328] @ wcat[11328,96] (WMMA) -> he_pre / hp_pre (- bias)
// ===========================================================================
__global__ void k_gemm2(const _Float16* __restrict__ h16, const _Float16* __restrict__ wcat,
                        const float* __restrict__ entry_b, const float* __restrict__ pool_b,
                        float* __restrict__ he_pre, float* __restrict__ hp_pre) {
  int mt = blockIdx.x;
  int tid = threadIdx.x;
  int nt = tid / 32, lane = tid % 32, hi = lane >> 4, ln = lane & 15;
  const _Float16* arow = h16 + (size_t)(mt * 16 + ln) * FDIM;
  v8f acc = {};
  for (int kc = 0; kc < FDIM; kc += 32) {
    __builtin_prefetch(arow + kc + 256, 0, 1);   // global_prefetch_b8
    v16h a, bm;
#pragma unroll
    for (int h = 0; h < 16; ++h) {
      a[h]  = arow[kc + akm(h, hi)];
      bm[h] = wcat[(kc + hi * 16 + h) * 96 + nt * 16 + ln];
    }
    acc = wmma16(a, bm, acc);
  }
  int j = nt * 16 + ln;
#pragma unroll
  for (int r = 0; r < 8; ++r) {
    int row = mt * 16 + r + hi * 8;
    int b = row / NC, node = row % NC;
    float v = acc[r];
    if (j < OGn) he_pre[(b * NC + node) * OGn + j] = v - entry_b[j];
    else         hp_pre[(b * NC + node) * PGn + (j - OGn)] = v - pool_b[j - OGn];
  }
}

// h_entry = per-node BN of he_pre
__global__ void k_henorm(const float* __restrict__ he_pre, const float* __restrict__ mhe,
                         const float* __restrict__ ishe, float* __restrict__ h_entry) {
  int total = NB * NC * OGn;
  for (int i = blockIdx.x * blockDim.x + threadIdx.x; i < total;
       i += gridDim.x * blockDim.x) {
    int node = (i / OGn) % NC;
    h_entry[i] = (he_pre[i] - mhe[node]) * ishe[node];
  }
}

// S10: h_pool BN -> s_assign = softmax(h_pool@pl_w + pl_b); entropy accumulated
__global__ void k_assign(const float* __restrict__ hp_pre, const float* __restrict__ mhp,
                         const float* __restrict__ ishp,
                         const float* __restrict__ pl_w, const float* __restrict__ pl_b,
                         float* __restrict__ s_assign, float* __restrict__ accs) {
  __shared__ float hp[PGn];
  __shared__ float sc[PGn];
  __shared__ float mx_se[2];
  int bn = blockIdx.x;
  int n = bn % NC;
  int tid = threadIdx.x;
  if (tid < PGn) hp[tid] = (hp_pre[bn * PGn + tid] - mhp[n]) * ishp[n];
  __syncthreads();
  if (tid < PGn) {
    float s = pl_b[tid];
    for (int k = 0; k < PGn; ++k) s += hp[k] * pl_w[k * PGn + tid];
    sc[tid] = s;
  }
  __syncthreads();
  if (tid == 0) {
    float mx = -1e30f;
    for (int j = 0; j < PGn; ++j) mx = fmaxf(mx, sc[j]);
    float se = 0.f;
    for (int j = 0; j < PGn; ++j) se += expf(sc[j] - mx);
    mx_se[0] = mx; mx_se[1] = se;
  }
  __syncthreads();
  if (tid < PGn) {
    float sa = expf(sc[tid] - mx_se[0]) / mx_se[1];
    s_assign[bn * PGn + tid] = sa;
    atomicAdd(&accs[1], -sa * logf(sa + 1e-30f));
  }
}

// ===========================================================================
// S11: per-batch dense_diff_pool + embed-GEMM pre-BN + rep1 + link-loss acc
// ===========================================================================
__global__ void k_diffpool(const float* __restrict__ s_assign,
                           const float* __restrict__ h_entry,
                           const float* __restrict__ adjf,
                           const float* __restrict__ embed_w,
                           const float* __restrict__ embed_b,
                           float* __restrict__ rep1, float* __restrict__ em_pre,
                           float* __restrict__ accs) {
  extern __shared__ float sm[];
  float* sA  = sm;                 // 62*32
  float* he  = sA + NC * PGn;      // 62*64
  float* ad  = he + NC * OGn;      // 62*62
  float* tmp = ad + NC * NC;       // 62*32
  float* xp  = tmp + NC * PGn;     // 32*64
  float* apl = xp + PGn * OGn;     // 32*32
  float* eb  = apl + PGn * PGn;    // 32*64
  float* red = eb + PGn * OGn;     // 256
  int b = blockIdx.x, tid = threadIdx.x;

  for (int i = tid; i < NC * PGn; i += blockDim.x) sA[i] = s_assign[b * NC * PGn + i];
  for (int i = tid; i < NC * OGn; i += blockDim.x) he[i] = h_entry[b * NC * OGn + i];
  for (int i = tid; i < NC * NC; i += blockDim.x) {
    int n = i / NC, m = i % NC;
    ad[i] = adjf[b * 4096 + n * 64 + m];
  }
  __syncthreads();
  // rep1 = sum_n h_entry (softmax over singleton axis == all ones)
  for (int f = tid; f < OGn; f += blockDim.x) {
    float s = 0.f;
    for (int n = 0; n < NC; ++n) s += he[n * OGn + f];
    rep1[b * OGn + f] = s;
  }
  // x_pool[k,f] = sum_n s[n,k] h_entry[n,f]
  for (int i = tid; i < PGn * OGn; i += blockDim.x) {
    int k = i / OGn, f = i % OGn;
    float s = 0.f;
    for (int n = 0; n < NC; ++n) s += sA[n * PGn + k] * he[n * OGn + f];
    xp[i] = s;
  }
  // tmp[n,l] = sum_m adj[n,m] s[m,l]
  for (int i = tid; i < NC * PGn; i += blockDim.x) {
    int n = i / PGn, l = i % PGn;
    float s = 0.f;
    for (int m = 0; m < NC; ++m) s += ad[n * NC + m] * sA[m * PGn + l];
    tmp[i] = s;
  }
  __syncthreads();
  // adj_pool[k,l] = sum_n s[n,k] tmp[n,l]
  for (int i = tid; i < PGn * PGn; i += blockDim.x) {
    int k = i / PGn, l = i % PGn;
    float s = 0.f;
    for (int n = 0; n < NC; ++n) s += sA[n * PGn + k] * tmp[n * PGn + l];
    apl[i] = s;
  }
  // link loss: ||adj - s s^T + eps||^2 partial sum
  float part = 0.f;
  for (int i = tid; i < NC * NC; i += blockDim.x) {
    int n = i / NC, m = i % NC;
    float ss = 0.f;
    for (int k = 0; k < PGn; ++k) ss += sA[n * PGn + k] * sA[m * PGn + k];
    float lk = ad[i] - ss + 1e-30f;
    part += lk * lk;
  }
  red[tid] = part;
  __syncthreads();
  for (int st = blockDim.x / 2; st > 0; st >>= 1) {
    if (tid < st) red[tid] += red[tid + st];
    __syncthreads();
  }
  if (tid == 0) atomicAdd(&accs[0], red[0]);
  // embed pre-BN: (adj_pool @ x_pool) @ embed_w - embed_b
  for (int i = tid; i < PGn * OGn; i += blockDim.x) {
    int k = i / OGn, f = i % OGn;
    float s = 0.f;
    for (int l = 0; l < PGn; ++l) s += apl[k * PGn + l] * xp[l * OGn + f];
    eb[i] = s;
  }
  __syncthreads();
  for (int i = tid; i < PGn * OGn; i += blockDim.x) {
    int k = i / OGn, j = i % OGn;
    float s = -embed_b[j];
    for (int f = 0; f < OGn; ++f) s += eb[k * OGn + f] * embed_w[f * OGn + j];
    em_pre[b * PGn * OGn + i] = s;
  }
}

// rep2 = sum_k BN(em_pre)
__global__ void k_rep2(const float* __restrict__ em_pre, const float* __restrict__ m2,
                       const float* __restrict__ is2, float* __restrict__ rep2) {
  int b = blockIdx.x;
  int f = threadIdx.x;  // 64
  float s = 0.f;
  for (int k = 0; k < PGn; ++k)
    s += (em_pre[(b * PGn + k) * OGn + f] - m2[k]) * is2[k];
  rep2[b * OGn + f] = s;
}

// ===========================================================================
// S13: head MLP + final BN over batch axis + loss outputs
// ===========================================================================
__global__ void k_head(const float* __restrict__ rep1, const float* __restrict__ rep2,
                       const float* __restrict__ fc1_w, const float* __restrict__ fc1_b,
                       const float* __restrict__ fc2_w, const float* __restrict__ fc2_b,
                       const float* __restrict__ fc3_w, const float* __restrict__ fc3_b,
                       const float* __restrict__ accs, float* __restrict__ outp) {
  __shared__ float o1[NB * 128];
  __shared__ float o2[NB * OGn];
  __shared__ float o3[NB * 4];
  int tid = threadIdx.x;
  for (int i = tid; i < NB * 128; i += blockDim.x) {
    int b = i / 128, j = i % 128;
    float s = fc1_b[j];
    for (int t = 0; t < OGn; ++t) s += rep1[b * OGn + t] * fc1_w[t * 128 + j];
    for (int t = 0; t < OGn; ++t) s += rep2[b * OGn + t] * fc1_w[(OGn + t) * 128 + j];
    o1[i] = fmaxf(s, 0.f);
  }
  __syncthreads();
  for (int i = tid; i < NB * OGn; i += blockDim.x) {
    int b = i / OGn, j = i % OGn;
    float s = fc2_b[j];
    for (int t = 0; t < 128; ++t) s += o1[b * 128 + t] * fc2_w[t * OGn + j];
    o2[i] = fmaxf(s, 0.f);
  }
  __syncthreads();
  for (int i = tid; i < NB * 4; i += blockDim.x) {
    int b = i / 4, j = i % 4;
    float s = fc3_b[j];
    for (int t = 0; t < OGn; ++t) s += o2[b * OGn + t] * fc3_w[t * 4 + j];
    o3[i] = s;
  }
  __syncthreads();
  if (tid < 4) {
    float m = 0.f, q = 0.f;
    for (int b = 0; b < NB; ++b) { float v = o3[b * 4 + tid]; m += v; q += v * v; }
    m *= (1.f / NB);
    q = q * (1.f / NB) - m * m;
    float is = rsqrtf(q + 1e-5f);
    for (int b = 0; b < NB; ++b) outp[b * 4 + tid] = (o3[b * 4 + tid] - m) * is;
  }
  if (tid == 0) {
    outp[NB * 4]     = sqrtf(accs[0]) / (float)(NB * NC * NC);  // link_loss
    outp[NB * 4 + 1] = accs[1] / (float)(NB * NC);              // ent_loss
  }
}

__global__ void k_init(float* accs) {
  if (threadIdx.x < 2) accs[threadIdx.x] = 0.f;
}

// ===========================================================================
extern "C" void kernel_launch(void* const* d_in, const int* in_sizes, int n_in,
                              void* d_out, int out_size, void* d_ws, size_t ws_size,
                              hipStream_t stream) {
  (void)in_sizes; (void)n_in; (void)out_size; (void)ws_size;
  const float* x       = (const float*)d_in[0];
  const float* conv1_w = (const float*)d_in[1];
  const float* conv1_b = (const float*)d_in[2];
  const float* conv2_w = (const float*)d_in[3];
  const float* conv2_b = (const float*)d_in[4];
  const float* conv3_w = (const float*)d_in[5];
  const float* conv3_b = (const float*)d_in[6];
  const float* oxo_w   = (const float*)d_in[7];
  const float* oxo_b   = (const float*)d_in[8];
  const float* lf_w    = (const float*)d_in[9];
  const float* lf_b    = (const float*)d_in[10];
  const float* gadj    = (const float*)d_in[11];
  const float* entry_w = (const float*)d_in[12];
  const float* entry_b = (const float*)d_in[13];
  const float* pool_w  = (const float*)d_in[14];
  const float* pool_b  = (const float*)d_in[15];
  const float* pl_w    = (const float*)d_in[16];
  const float* pl_b    = (const float*)d_in[17];
  const float* embed_w = (const float*)d_in[18];
  const float* embed_b = (const float*)d_in[19];
  const float* fc1_w   = (const float*)d_in[20];
  const float* fc1_b   = (const float*)d_in[21];
  const float* fc2_w   = (const float*)d_in[22];
  const float* fc2_b   = (const float*)d_in[23];
  const float* fc3_w   = (const float*)d_in[24];
  const float* fc3_b   = (const float*)d_in[25];
  float* outp = (float*)d_out;

  char* ws = (char*)d_ws;
  size_t off = 0;
  auto carve = [&](size_t bytes) -> char* {
    char* p = ws + off;
    off = (off + bytes + 255) & ~(size_t)255;
    return p;
  };
  float*     out0     = (float*)carve((size_t)NB * NTF * NC * LTOT * 4);
  float*     m0       = (float*)carve(NTF * 4);
  float*     is0      = (float*)carve(NTF * 4);
  float*     out1     = (float*)carve((size_t)NB * NTF * NC * LHALF * 4);
  float*     m1       = (float*)carve(NTF * 4);
  float*     is1      = (float*)carve(NTF * 4);
  float*     feat_pre = (float*)carve((size_t)NB * NC * FDIM * 4);
  float*     mc       = (float*)carve(NC * 4);
  float*     isc      = (float*)carve(NC * 4);
  _Float16*  feat_h   = (_Float16*)carve((size_t)NB * 64 * FDIM * 2);
  float*     adjf     = (float*)carve((size_t)NB * 64 * 64 * 4);
  _Float16*  adjh     = (_Float16*)carve((size_t)NB * 64 * 64 * 2);
  _Float16*  h16      = (_Float16*)carve((size_t)NB * NC * FDIM * 2);
  _Float16*  wcat     = (_Float16*)carve((size_t)FDIM * 96 * 2);
  float*     he_pre   = (float*)carve((size_t)NB * NC * OGn * 4);
  float*     hp_pre   = (float*)carve((size_t)NB * NC * PGn * 4);
  float*     mhe      = (float*)carve(NC * 4);
  float*     ishe     = (float*)carve(NC * 4);
  float*     mhp      = (float*)carve(NC * 4);
  float*     ishp     = (float*)carve(NC * 4);
  float*     h_entry  = (float*)carve((size_t)NB * NC * OGn * 4);
  float*     s_assign = (float*)carve((size_t)NB * NC * PGn * 4);
  float*     rep1     = (float*)carve(NB * OGn * 4);
  float*     rep2     = (float*)carve(NB * OGn * 4);
  float*     em_pre   = (float*)carve((size_t)NB * PGn * OGn * 4);
  float*     m2       = (float*)carve(PGn * 4);
  float*     is2      = (float*)carve(PGn * 4);
  float*     accs     = (float*)carve(2 * 4);

  const int CONV_LDS = (1152 + 32 * 128) * 2 + (32 + 32 * 1008) * 4;        // 139648
  const int OXO_LDS  = (32 * 720 + 32 * 32) * 2 + (32 + 32 * 712) * 4;      // 139392
  const int DIFF_LDS = (NC*PGn + NC*OGn + NC*NC + NC*PGn + PGn*OGn +
                        PGn*PGn + PGn*OGn + 256) * 4;                       // 68624
  (void)hipFuncSetAttribute((const void*)k_conv,
        hipFuncAttributeMaxDynamicSharedMemorySize, CONV_LDS);
  (void)hipFuncSetAttribute((const void*)k_oxo,
        hipFuncAttributeMaxDynamicSharedMemorySize, OXO_LDS);
  (void)hipFuncSetAttribute((const void*)k_diffpool,
        hipFuncAttributeMaxDynamicSharedMemorySize, DIFF_LDS);

  k_init<<<1, 32, 0, stream>>>(accs);
  {
    dim3 g(NB * NC, 3);
    k_conv<<<g, 256, CONV_LDS, stream>>>(x, conv1_w, conv1_b, conv2_w, conv2_b,
                                         conv3_w, conv3_b, out0);
  }
  k_stats<<<NTF, 256, 0, stream>>>(out0, NB, NTF, NC * LTOT, m0, is0);
  k_oxo<<<NB * NC, 256, OXO_LDS, stream>>>(out0, oxo_w, oxo_b, m0, is0, out1);
  k_stats<<<NTF, 256, 0, stream>>>(out1, NB, NTF, NC * LHALF, m1, is1);
  k_feat<<<NB * NC, 256, 0, stream>>>(out1, m1, is1, lf_w, lf_b, feat_pre);
  k_stats<<<NC, 256, 0, stream>>>(feat_pre, NB, NC, FDIM, mc, isc);
  k_featnorm<<<2048, 256, 0, stream>>>(feat_pre, mc, isc, feat_h);
  k_adj<<<NB, 512, 0, stream>>>(feat_h, gadj, adjf, adjh);
  {
    dim3 g(FDIM / 16, NB);
    k_hpre<<<g, 128, 0, stream>>>(adjh, feat_h, h16);
  }
  k_wcat<<<1024, 256, 0, stream>>>(entry_w, pool_w, wcat);
  k_gemm2<<<(NB * NC) / 16, 192, 0, stream>>>(h16, wcat, entry_b, pool_b,
                                              he_pre, hp_pre);
  k_stats<<<NC, 256, 0, stream>>>(he_pre, NB, NC, OGn, mhe, ishe);
  k_stats<<<NC, 256, 0, stream>>>(hp_pre, NB, NC, PGn, mhp, ishp);
  k_henorm<<<512, 256, 0, stream>>>(he_pre, mhe, ishe, h_entry);
  k_assign<<<NB * NC, 64, 0, stream>>>(hp_pre, mhp, ishp, pl_w, pl_b, s_assign, accs);
  k_diffpool<<<NB, 256, DIFF_LDS, stream>>>(s_assign, h_entry, adjf,
                                            embed_w, embed_b, rep1, em_pre, accs);
  k_stats<<<PGn, 256, 0, stream>>>(em_pre, NB, PGn, OGn, m2, is2);
  k_rep2<<<NB, OGn, 0, stream>>>(em_pre, m2, is2, rep2);
  k_head<<<1, 256, 0, stream>>>(rep1, rep2, fc1_w, fc1_b, fc2_w, fc2_b,
                                fc3_w, fc3_b, accs, outp);
}
